// BigBirdAttentionAblation_13511967113471
// MI455X (gfx1250) — matrix operand order repaired and checked by reading, not compiled, not run on hardware
//
#include <hip/hip_runtime.h>
#include <hip/hip_bf16.h>

typedef _Float16 v8h  __attribute__((ext_vector_type(8)));
typedef _Float16 v16h __attribute__((ext_vector_type(16)));
typedef float    v4f  __attribute__((ext_vector_type(4)));
typedef float    v8f  __attribute__((ext_vector_type(8)));

#define DIMX   1536
#define NH     8
#define DKH    64
#define DVH    192
#define BSQ    128
#define NBLK   32
#define NSEQ   4096
#define NBATCH 2
#define MTOT   8192   // NBATCH*NSEQ

static __device__ inline v8f zero8() {
  v8f z;
#pragma unroll
  for (int i = 0; i < 8; ++i) z[i] = 0.0f;
  return z;
}

// Low 32 bits of a generic pointer into __shared__ are the LDS byte offset.
static __device__ inline unsigned lds_off(const void* p) {
  return (unsigned)(unsigned long long)p;
}

// Async 16B copy global -> LDS (GVS mode: saddr base + 32-bit voffset).
static __device__ inline void async_b128(unsigned ldsa, unsigned goff,
                                         const void* base) {
  asm volatile("global_load_async_to_lds_b128 %0, %1, %2"
               :: "v"(ldsa), "v"(goff), "s"(base)
               : "memory");
}
static __device__ inline void wait_async0() {
  asm volatile("s_wait_asynccnt 0" ::: "memory");
}

// Build a 16-half fragment from two 16B-aligned LDS slices (8+8 halves).
static __device__ inline v16h frag_ab(const _Float16* lo, const _Float16* hi) {
  v8h l = *(const v8h*)lo;
  v8h h = *(const v8h*)hi;
  v16h r;
#pragma unroll
  for (int i = 0; i < 8; ++i) { r[i] = l[i]; r[i + 8] = h[i]; }
  return r;
}

static __device__ inline v8f wmma16(v16h a, v16h b, v8f c) {
  // D = A(16x32 f16) x B(32x16 f16) + C(16x16 f32)
  return __builtin_amdgcn_wmma_f32_16x16x32_f16(false, a, false, b, (short)0, c,
                                                false, false);
}

// ---------------------------------------------------------------------------
// GEMMs: block tile 128x128, K-step 32. 8 waves; wave w owns M strips
// {2*(w&3), 2*(w&3)+1} and N strips {(w>>2)*4 .. +3}: 8 WMMAs per K-step,
// 2 A-frags + 4 B-frags (12 ds_load_b128).
// ---------------------------------------------------------------------------
#define GK 32
#define GM 128
#define GN 128
#define AS_STR 40   // halves per row (32 + 8 pad), 16B aligned
#define BS_STR 40

// C(f16)[M,N] = A(f32)[M,K] * B(f32)[K,N]
__global__ __launch_bounds__(256) void gemm_f32f16(
    const float* __restrict__ A, const float* __restrict__ B,
    _Float16* __restrict__ C, int K, int N) {
  __shared__ __align__(16) _Float16 As[GM * AS_STR];
  __shared__ __align__(16) _Float16 Bs[GN * BS_STR];  // transposed: Bs[n][k]
  const int t    = threadIdx.x;
  const int lane = t & 31;
  const int w    = t >> 5;
  const int lm   = lane & 15;
  const int half8 = (lane < 16) ? 0 : 8;
  const int koffB = (lane < 16) ? 0 : 16;
  const int n0 = blockIdx.x * GN;
  const int m0 = blockIdx.y * GM;
  const int ms0 = 2 * (w & 3);        // first M strip of this wave
  const int ns0 = (w >> 2) * 4;       // first N strip of this wave

  v8f acc[2][4];
#pragma unroll
  for (int mi = 0; mi < 2; ++mi)
#pragma unroll
    for (int j = 0; j < 4; ++j) acc[mi][j] = zero8();

  const int ar = t >> 1;            // A stage: row 0..127
  const int ak = (t & 1) * 16;      // k offset 0/16
  const int bk = t >> 3;            // B stage: k row 0..31
  const int bn = (t & 7) * 16;      // n offset 0..112

  for (int kb = 0; kb < K; kb += GK) {
    __syncthreads();
    {  // stage A (f32 -> f16)
      const float* ap = A + (size_t)(m0 + ar) * K + kb + ak;
      v4f a0 = *(const v4f*)(ap + 0);
      v4f a1 = *(const v4f*)(ap + 4);
      v4f a2 = *(const v4f*)(ap + 8);
      v4f a3 = *(const v4f*)(ap + 12);
      _Float16* dst = &As[ar * AS_STR + ak];
#pragma unroll
      for (int i = 0; i < 4; ++i) {
        dst[i]      = (_Float16)a0[i];
        dst[4 + i]  = (_Float16)a1[i];
        dst[8 + i]  = (_Float16)a2[i];
        dst[12 + i] = (_Float16)a3[i];
      }
    }
    {  // stage B transposed (f32 -> f16)
      const float* bp = B + (size_t)(kb + bk) * N + n0 + bn;
      v4f b0 = *(const v4f*)(bp + 0);
      v4f b1 = *(const v4f*)(bp + 4);
      v4f b2 = *(const v4f*)(bp + 8);
      v4f b3 = *(const v4f*)(bp + 12);
#pragma unroll
      for (int i = 0; i < 4; ++i) {
        Bs[(bn + i) * BS_STR + bk]      = (_Float16)b0[i];
        Bs[(bn + 4 + i) * BS_STR + bk]  = (_Float16)b1[i];
        Bs[(bn + 8 + i) * BS_STR + bk]  = (_Float16)b2[i];
        Bs[(bn + 12 + i) * BS_STR + bk] = (_Float16)b3[i];
      }
    }
    __syncthreads();
    v16h af[2];
#pragma unroll
    for (int mi = 0; mi < 2; ++mi) {
      const int arow = (ms0 + mi) * 16 + lm;
      af[mi] = frag_ab(&As[arow * AS_STR + half8],
                       &As[arow * AS_STR + 16 + half8]);
    }
#pragma unroll
    for (int jt = 0; jt < 4; ++jt) {
      const int brow = (ns0 + jt) * 16 + lm;
      v16h bf = frag_ab(&Bs[brow * BS_STR + half8],
                        &Bs[brow * BS_STR + 16 + half8]);
      acc[0][jt] = wmma16(af[0], bf, acc[0][jt]);
      acc[1][jt] = wmma16(af[1], bf, acc[1][jt]);
    }
  }
#pragma unroll
  for (int mi = 0; mi < 2; ++mi)
#pragma unroll
    for (int jt = 0; jt < 4; ++jt)
#pragma unroll
      for (int i = 0; i < 8; ++i) {
        const int row = m0 + (ms0 + mi) * 16 + i + half8;
        const int col = n0 + (ns0 + jt) * 16 + lm;
        C[(size_t)row * N + col] = (_Float16)acc[mi][jt][i];
      }
}

// C(f32)[M,N] = A(f16)[M,K] * B(f32)[K,N] + bias[N]; A staged via async copy.
__global__ __launch_bounds__(256) void gemm_f16f32_bias(
    const _Float16* __restrict__ A, const float* __restrict__ B,
    const float* __restrict__ bias, float* __restrict__ C, int K, int N) {
  __shared__ __align__(16) _Float16 As[GM * AS_STR];
  __shared__ __align__(16) _Float16 Bs[GN * BS_STR];
  const int t    = threadIdx.x;
  const int lane = t & 31;
  const int w    = t >> 5;
  const int lm   = lane & 15;
  const int half8 = (lane < 16) ? 0 : 8;
  const int koffB = (lane < 16) ? 0 : 16;
  const int n0 = blockIdx.x * GN;
  const int m0 = blockIdx.y * GM;
  const int ms0 = 2 * (w & 3);
  const int ns0 = (w >> 2) * 4;

  v8f acc[2][4];
#pragma unroll
  for (int mi = 0; mi < 2; ++mi)
#pragma unroll
    for (int j = 0; j < 4; ++j) acc[mi][j] = zero8();

  const int ar = t >> 1;
  const int ak = (t & 1) * 16;
  const int bk = t >> 3;
  const int bn = (t & 7) * 16;
  const unsigned aldsa = lds_off(&As[ar * AS_STR + ak]);

  for (int kb = 0; kb < K; kb += GK) {
    __syncthreads();
    {  // stage A (f16) via async global->LDS, 2x16B per thread
      const unsigned goff = (unsigned)(((size_t)(m0 + ar) * K + kb + ak) * 2);
      async_b128(aldsa, goff, A);
      async_b128(aldsa + 16, goff + 16, A);
    }
    {  // stage B transposed (f32 -> f16)
      const float* bp = B + (size_t)(kb + bk) * N + n0 + bn;
      v4f b0 = *(const v4f*)(bp + 0);
      v4f b1 = *(const v4f*)(bp + 4);
      v4f b2 = *(const v4f*)(bp + 8);
      v4f b3 = *(const v4f*)(bp + 12);
#pragma unroll
      for (int i = 0; i < 4; ++i) {
        Bs[(bn + i) * BS_STR + bk]      = (_Float16)b0[i];
        Bs[(bn + 4 + i) * BS_STR + bk]  = (_Float16)b1[i];
        Bs[(bn + 8 + i) * BS_STR + bk]  = (_Float16)b2[i];
        Bs[(bn + 12 + i) * BS_STR + bk] = (_Float16)b3[i];
      }
    }
    wait_async0();
    __syncthreads();
    v16h af[2];
#pragma unroll
    for (int mi = 0; mi < 2; ++mi) {
      const int arow = (ms0 + mi) * 16 + lm;
      af[mi] = frag_ab(&As[arow * AS_STR + half8],
                       &As[arow * AS_STR + 16 + half8]);
    }
#pragma unroll
    for (int jt = 0; jt < 4; ++jt) {
      const int brow = (ns0 + jt) * 16 + lm;
      v16h bf = frag_ab(&Bs[brow * BS_STR + half8],
                        &Bs[brow * BS_STR + 16 + half8]);
      acc[0][jt] = wmma16(af[0], bf, acc[0][jt]);
      acc[1][jt] = wmma16(af[1], bf, acc[1][jt]);
    }
  }
#pragma unroll
  for (int jt = 0; jt < 4; ++jt) {
    const int col = n0 + (ns0 + jt) * 16 + lm;
    const float bb = bias[col];
#pragma unroll
    for (int mi = 0; mi < 2; ++mi)
#pragma unroll
      for (int i = 0; i < 8; ++i) {
        const int row = m0 + (ms0 + mi) * 16 + i + half8;
        C[(size_t)row * N + col] = acc[mi][jt][i] + bb;
      }
  }
}

// ---------------------------------------------------------------------------
// Block-local flash attention. Grid (H, NBLK, NBATCH), 256 threads (8 waves).
// Wave w owns query rows 16w..16w+15. Keys: 3-block window, chunks of 32,
// OOB keys staged as zeros (participate in softmax, matching the reference).
// S^T = K*Q^T so softmax stats are per-lane; exp'd S^T accumulators are
// directly the A-fragment of P for the P@V WMMA. Q/K staged with async copies.
// ---------------------------------------------------------------------------
#define QS_STR 72   // 64 + 8 pad halves
#define KS_STR 72
#define VT_STR 40   // 32 + 8 pad halves

__global__ __launch_bounds__(256) void attn_kernel(
    const _Float16* __restrict__ Qg, const _Float16* __restrict__ Kg,
    const _Float16* __restrict__ Vg, _Float16* __restrict__ Og) {
  __shared__ __align__(16) _Float16 Qs[BSQ * QS_STR];
  __shared__ __align__(16) _Float16 Ks[32 * KS_STR];
  __shared__ __align__(16) _Float16 VTs[DVH * VT_STR];  // transposed [dv][key]

  const int h   = blockIdx.x;
  const int blk = blockIdx.y;
  const int b   = blockIdx.z;
  const int t    = threadIdx.x;
  const int lane = t & 31;
  const int w    = t >> 5;
  const int lm   = lane & 15;
  const int half8 = (lane < 16) ? 0 : 8;
  const int koffB = (lane < 16) ? 0 : 16;

  {  // stage Q tile [128][64] via async copy: 4x16B per thread
    const int row = t >> 1;
    const int d0  = (t & 1) * 32;
    const unsigned goff = (unsigned)(
        ((b * NSEQ + blk * BSQ + row) * (NH * DKH) + h * DKH + d0) * 2);
    const unsigned ldsa = lds_off(&Qs[row * QS_STR + d0]);
#pragma unroll
    for (int i = 0; i < 4; ++i) async_b128(ldsa + 16 * i, goff + 16 * i, Qg);
  }
  wait_async0();
  __syncthreads();

  // Q fragments (B-layout: lane = query row, d = 8h+s / 16+8h+(s-8) per half)
  const int qrow = w * 16 + lm;
  v16h qf0 = frag_ab(&Qs[qrow * QS_STR + half8], &Qs[qrow * QS_STR + 16 + half8]);
  v16h qf1 = frag_ab(&Qs[qrow * QS_STR + 32 + half8],
                     &Qs[qrow * QS_STR + 48 + half8]);

  v8f o[12];
#pragma unroll
  for (int i = 0; i < 12; ++i) o[i] = zero8();
  float mrun = -1e30f, lrun = 0.0f;

  const int key_base = blk * BSQ - BSQ;  // window start (may be < 0)
  const int krow = t >> 3;               // 0..31
  const int kd0  = (t & 7) * 8;          // 0..56
  const int vd0  = (t & 7) * 24;         // 0..168
  const unsigned kldsa = lds_off(&Ks[krow * KS_STR + kd0]);

  for (int c = 0; c < 12; ++c) {
    __syncthreads();
    {  // stage K chunk [32][64]: async for valid keys, ds-store zeros for OOB
      const int key = key_base + c * 32 + krow;
      if (key >= 0 && key < NSEQ) {
        const unsigned goff =
            (unsigned)(((b * NSEQ + key) * (NH * DKH) + h * DKH + kd0) * 2);
        async_b128(kldsa, goff, Kg);
      } else {
        v8h z;
#pragma unroll
        for (int i = 0; i < 8; ++i) z[i] = (_Float16)0.0f;
        *(v8h*)&Ks[krow * KS_STR + kd0] = z;
      }
    }
    {  // stage V chunk transposed: VTs[dv][key]
      const int key = key_base + c * 32 + krow;
      if (key >= 0 && key < NSEQ) {
        const _Float16* vp =
            Vg + (size_t)(b * NSEQ + key) * (NH * DVH) + h * DVH + vd0;
        v8h x0 = *(const v8h*)(vp + 0);
        v8h x1 = *(const v8h*)(vp + 8);
        v8h x2 = *(const v8h*)(vp + 16);
#pragma unroll
        for (int i = 0; i < 8; ++i) {
          VTs[(vd0 + i) * VT_STR + krow]      = x0[i];
          VTs[(vd0 + 8 + i) * VT_STR + krow]  = x1[i];
          VTs[(vd0 + 16 + i) * VT_STR + krow] = x2[i];
        }
      } else {
#pragma unroll
        for (int i = 0; i < 24; ++i)
          VTs[(vd0 + i) * VT_STR + krow] = (_Float16)0.0f;
      }
    }
    wait_async0();
    __syncthreads();

    // S^T tiles: D[key][qrow]; A = K rows (M=key), B = Q^T (N=qrow)
    v8f st0 = zero8(), st1 = zero8();
    {
      v16h ka = frag_ab(&Ks[lm * KS_STR + half8], &Ks[lm * KS_STR + 16 + half8]);
      v16h kb2 = frag_ab(&Ks[lm * KS_STR + 32 + half8],
                         &Ks[lm * KS_STR + 48 + half8]);
      st0 = wmma16(ka, qf0, st0);
      st0 = wmma16(kb2, qf1, st0);
      const int r1 = 16 + lm;
      v16h kc = frag_ab(&Ks[r1 * KS_STR + half8], &Ks[r1 * KS_STR + 16 + half8]);
      v16h kd = frag_ab(&Ks[r1 * KS_STR + 32 + half8],
                        &Ks[r1 * KS_STR + 48 + half8]);
      st1 = wmma16(kc, qf0, st1);
      st1 = wmma16(kd, qf1, st1);
    }

    // online softmax; each lane holds 16 of its row's 32 chunk scores,
    // the other 16 live in lane^16
    float cmax = -1e30f;
#pragma unroll
    for (int i = 0; i < 8; ++i) {
      st0[i] *= 0.125f;  // 1/sqrt(DK)
      st1[i] *= 0.125f;
      cmax = fmaxf(cmax, fmaxf(st0[i], st1[i]));
    }
    cmax = fmaxf(cmax, __shfl_xor(cmax, 16, 32));
    const float mnew  = fmaxf(mrun, cmax);
    const float alpha = __expf(mrun - mnew);
    float p0[8], p1[8], rs = 0.0f;
#pragma unroll
    for (int i = 0; i < 8; ++i) {
      p0[i] = __expf(st0[i] - mnew);
      p1[i] = __expf(st1[i] - mnew);
      rs += p0[i] + p1[i];
    }
    rs += __shfl_xor(rs, 16, 32);
    lrun = lrun * alpha + rs;
    mrun = mnew;

    // rescale O by per-row alpha (row of O tile i is qrow i+half8)
    float arow[8];
#pragma unroll
    for (int i = 0; i < 8; ++i) arow[i] = __shfl(alpha, i + half8, 32);
#pragma unroll
    for (int tI = 0; tI < 12; ++tI)
#pragma unroll
      for (int i = 0; i < 8; ++i) o[tI][i] *= arow[i];

    // P A-fragment: exp'd S^T accumulators map 1:1 into A layout per lane
    v16h pf;
#pragma unroll
    for (int i = 0; i < 8; ++i) {
      pf[i]     = (_Float16)p0[i];
      pf[8 + i] = (_Float16)p1[i];
    }
    // O += P @ V  (12 dv tiles)
#pragma unroll
    for (int tI = 0; tI < 12; ++tI) {
      const int dv = tI * 16 + lm;
      v16h vf = frag_ab(&VTs[dv * VT_STR + half8], &VTs[dv * VT_STR + 16 + half8]);
      o[tI] = wmma16(pf, vf, o[tI]);
    }
  }

  // normalize and store attn output (f16)
  const float linv = 1.0f / lrun;
  float lrow[8];
#pragma unroll
  for (int i = 0; i < 8; ++i) lrow[i] = __shfl(linv, i + half8, 32);
  const size_t rowbase = (size_t)(b * NSEQ + blk * BSQ + w * 16);
#pragma unroll
  for (int tI = 0; tI < 12; ++tI)
#pragma unroll
    for (int i = 0; i < 8; ++i) {
      const size_t row = rowbase + i + half8;
      const int col = h * DVH + tI * 16 + lm;
      Og[row * (NH * DVH) + col] = (_Float16)(o[tI][i] * lrow[i]);
    }
}

// ---------------------------------------------------------------------------
extern "C" void kernel_launch(void* const* d_in, const int* in_sizes, int n_in,
                              void* d_out, int out_size, void* d_ws,
                              size_t ws_size, hipStream_t stream) {
  const float* x  = (const float*)d_in[0];
  const float* Wq = (const float*)d_in[1];
  const float* Wk = (const float*)d_in[2];
  const float* Wv = (const float*)d_in[3];
  const float* Wo = (const float*)d_in[4];
  const float* bo = (const float*)d_in[5];
  float* out = (float*)d_out;

  // f16 workspace: Q[8192,512] K[8192,512] V[8192,1536] attn[8192,1536]
  _Float16* Qh = (_Float16*)d_ws;
  _Float16* Kh = Qh + (size_t)MTOT * (NH * DKH);
  _Float16* Vh = Kh + (size_t)MTOT * (NH * DKH);
  _Float16* Ah = Vh + (size_t)MTOT * (NH * DVH);

  dim3 blkT(256);
  // projections
  gemm_f32f16<<<dim3((NH * DKH) / GN, MTOT / GM), blkT, 0, stream>>>(
      x, Wq, Qh, DIMX, NH * DKH);
  gemm_f32f16<<<dim3((NH * DKH) / GN, MTOT / GM), blkT, 0, stream>>>(
      x, Wk, Kh, DIMX, NH * DKH);
  gemm_f32f16<<<dim3((NH * DVH) / GN, MTOT / GM), blkT, 0, stream>>>(
      x, Wv, Vh, DIMX, NH * DVH);
  // sliding-window attention
  attn_kernel<<<dim3(NH, NBLK, NBATCH), blkT, 0, stream>>>(Qh, Kh, Vh, Ah);
  // output projection + bias
  gemm_f16f32_bias<<<dim3(DIMX / GN, MTOT / GM), blkT, 0, stream>>>(
      Ah, Wo, bo, out, NH * DVH, DIMX);
}
